// MiniGPT_48601849921567
// MI455X (gfx1250) — matrix operand order repaired and checked by reading, not compiled, not run on hardware
//
#include <hip/hip_runtime.h>
#include <cstdint>
#include <cstddef>

// ---------------- model dims (compile-time, from reference) ----------------
#define TSEQ   2048
#define NBATCH 2
#define CDIM   512
#define NH     8
#define HD     64
#define NL     6
#define VOC    32000
#define MROWS  (NBATCH * TSEQ)   // 4096 token rows

typedef __attribute__((ext_vector_type(16))) _Float16 v16h;
typedef __attribute__((ext_vector_type(8)))  float    v8f;

union HF16 { v16h v; uint4 q[2]; };

__device__ __forceinline__ v8f v8f_zero() {
  v8f z;
#pragma unroll
  for (int i = 0; i < 8; ++i) z[i] = 0.0f;
  return z;
}

// A-matrix 16x32 f16 fragment: lane-lo holds K {0..7,16..23}, lane-hi {8..15,24..31}
__device__ __forceinline__ v16h frag_a16(const _Float16* rowp, int hi) {
  HF16 f; const int b = hi ? 8 : 0;
  f.q[0] = *(const uint4*)(rowp + b);
  f.q[1] = *(const uint4*)(rowp + b + 16);
  return f.v;
}
// B-matrix 32x16 f16 fragment: lane-lo holds K {0..7,16..23}, lane-hi {8..15,24..31}
__device__ __forceinline__ v16h frag_b16(const _Float16* colp, int hi) {
  HF16 f; const int b = hi ? 8 : 0;
  f.q[0] = *(const uint4*)(colp + b);
  f.q[1] = *(const uint4*)(colp + b + 16);
  return f.v;
}

__device__ __forceinline__ void cvt16f_h(const float* __restrict__ src,
                                         _Float16* __restrict__ dst) {
#pragma unroll
  for (int i = 0; i < 16; i += 4) {
    const float4 f = *(const float4*)(src + i);
    dst[i + 0] = (_Float16)f.x; dst[i + 1] = (_Float16)f.y;
    dst[i + 2] = (_Float16)f.z; dst[i + 3] = (_Float16)f.w;
  }
}

// ---------------------------------------------------------------------------
// Embedding gather: x[row, :] = tok_embed[idx[row], :] + pos_embed[row % T, :]
// ---------------------------------------------------------------------------
__global__ __launch_bounds__(256) void embed_kernel(
    const int* __restrict__ idx, const float* __restrict__ tok,
    const float* __restrict__ pos, float* __restrict__ x)
{
  const int row = blockIdx.x, tid = threadIdx.x;
  const int token = idx[row];
  const int t = row & (TSEQ - 1);
  x[(size_t)row * CDIM + tid] =
      tok[(size_t)token * CDIM + tid] + pos[(size_t)t * CDIM + tid];
  x[(size_t)row * CDIM + tid + 256] =
      tok[(size_t)token * CDIM + tid + 256] + pos[(size_t)t * CDIM + tid + 256];
}

// ---------------------------------------------------------------------------
// LayerNorm (fp32 in) -> f16 out  (one 256-thread block per row of 512)
// ---------------------------------------------------------------------------
__global__ __launch_bounds__(256) void ln_kernel(
    const float* __restrict__ x, const float* __restrict__ w,
    const float* __restrict__ b, _Float16* __restrict__ out)
{
  __shared__ float red[256];
  const int row = blockIdx.x, tid = threadIdx.x;
  const float* xr = x + (size_t)row * CDIM;
  const float v0 = xr[tid], v1 = xr[tid + 256];
  red[tid] = v0 + v1; __syncthreads();
  for (int s = 128; s > 0; s >>= 1) { if (tid < s) red[tid] += red[tid + s]; __syncthreads(); }
  const float mu = red[0] * (1.0f / CDIM); __syncthreads();
  const float d0 = v0 - mu, d1 = v1 - mu;
  red[tid] = d0 * d0 + d1 * d1; __syncthreads();
  for (int s = 128; s > 0; s >>= 1) { if (tid < s) red[tid] += red[tid + s]; __syncthreads(); }
  const float inv = rsqrtf(red[0] * (1.0f / CDIM) + 1e-5f);
  out[(size_t)row * CDIM + tid]       = (_Float16)(d0 * inv * w[tid] + b[tid]);
  out[(size_t)row * CDIM + tid + 256] = (_Float16)(d1 * inv * w[tid + 256] + b[tid + 256]);
}

// ---------------------------------------------------------------------------
// WMMA GEMM:  out[M,N] = epilogue( A_f16[M,K] @ W_f32[N,K]^T + bias[N] )
// Block tile 128x128, BK=32, 256 threads (8 waves), wave tile 32x64 (2x4 tiles)
// Software-pipelined: double-buffered LDS (one barrier per K-step), with
// global_prefetch of the K+2 tile to warm L2/WGP$.
// ---------------------------------------------------------------------------
template<bool DOGELU, bool DORESID, bool OUT16>
__global__ __launch_bounds__(256) void gemm_kernel(
    const _Float16* __restrict__ A, const float* __restrict__ W,
    const float* __restrict__ bias, void* __restrict__ outv,
    const float* __restrict__ resid, int M, int N, int K)
{
  __shared__ __align__(16) _Float16 ldsA[2][128 * 32];
  __shared__ __align__(16) _Float16 ldsB[2][128 * 32];
  const int tid  = threadIdx.x;
  const int lane = tid & 31;
  const int wv   = tid >> 5;
  const int hi   = lane >> 4;
  const int l15  = lane & 15;
  const int m0 = blockIdx.y * 128;
  const int n0 = blockIdx.x * 128;
  const int wm = (wv >> 1) * 32;   // wave row offset: 0,32,64,96
  const int wn = (wv & 1) * 64;    // wave col offset: 0,64

  v8f acc[2][4];
#pragma unroll
  for (int i = 0; i < 2; ++i)
#pragma unroll
    for (int j = 0; j < 4; ++j) acc[i][j] = v8f_zero();

  const int lr = tid >> 1;         // 0..127: tile row loaded by this thread
  const int lk = (tid & 1) * 16;   // 0 or 16: k-half
  const _Float16* gA = A + (size_t)(m0 + lr) * K + lk;
  const float*    gW = W + (size_t)(n0 + lr) * K + lk;

  // prologue: fill buffer 0 with tile 0
  {
    const uint4 a0 = *(const uint4*)(gA);
    const uint4 a1 = *(const uint4*)(gA + 8);
    __align__(16) _Float16 tb[16];
    cvt16f_h(gW, tb);
    *(uint4*)&ldsA[0][lr * 32 + lk]     = a0;
    *(uint4*)&ldsA[0][lr * 32 + lk + 8] = a1;
    *(uint4*)&ldsB[0][lr * 32 + lk]     = *(const uint4*)&tb[0];
    *(uint4*)&ldsB[0][lr * 32 + lk + 8] = *(const uint4*)&tb[8];
  }
  __syncthreads();

  const int S = K >> 5;            // K / 32 steps
  for (int s = 0; s < S; ++s) {
    const int cur = s & 1, nxt = cur ^ 1;
    const bool hasNext = (s + 1) < S;
    uint4 a0{}, a1{};
    __align__(16) _Float16 tb[16];
    if (hasNext) {                 // issue next tile's global loads early
      const _Float16* pa = gA + (size_t)(s + 1) * 32;
      const float*    pw = gW + (size_t)(s + 1) * 32;
      a0 = *(const uint4*)(pa);
      a1 = *(const uint4*)(pa + 8);
      cvt16f_h(pw, tb);
      if (s + 2 < S) {             // warm L2 for the tile after next
        __builtin_prefetch(gA + (size_t)(s + 2) * 32, 0, 1);
        __builtin_prefetch(gW + (size_t)(s + 2) * 32, 0, 1);
      }
    }
    // consume current LDS buffer
    v16h af[2], bf[4];
#pragma unroll
    for (int i = 0; i < 2; ++i) af[i] = frag_a16(&ldsA[cur][(wm + i * 16 + l15) * 32], hi);
#pragma unroll
    for (int j = 0; j < 4; ++j) bf[j] = frag_b16(&ldsB[cur][(wn + j * 16 + l15) * 32], hi);
#pragma unroll
    for (int i = 0; i < 2; ++i)
#pragma unroll
      for (int j = 0; j < 4; ++j)
        acc[i][j] = __builtin_amdgcn_wmma_f32_16x16x32_f16(
            false, af[i], false, bf[j], (short)0, acc[i][j], false, false);
    // stage next tile into the other buffer; single barrier per step
    if (hasNext) {
      *(uint4*)&ldsA[nxt][lr * 32 + lk]     = a0;
      *(uint4*)&ldsA[nxt][lr * 32 + lk + 8] = a1;
      *(uint4*)&ldsB[nxt][lr * 32 + lk]     = *(const uint4*)&tb[0];
      *(uint4*)&ldsB[nxt][lr * 32 + lk + 8] = *(const uint4*)&tb[8];
    }
    __syncthreads();
  }

  // epilogue: C layout — VGPR r, lane-lo -> (m=r, n=l15), lane-hi -> (m=r+8, n=l15)
#pragma unroll
  for (int i = 0; i < 2; ++i)
#pragma unroll
    for (int j = 0; j < 4; ++j) {
      const int n = n0 + wn + j * 16 + l15;
      const float bv = bias[n];
#pragma unroll
      for (int r = 0; r < 8; ++r) {
        const int m = m0 + wm + i * 16 + r + (hi ? 8 : 0);
        float v = acc[i][j][r] + bv;
        if (DOGELU) v = 0.5f * v * (1.0f + erff(v * 0.70710678118654752f));
        if (DORESID) v += resid[(size_t)m * N + n];
        if (OUT16) ((_Float16*)outv)[(size_t)m * N + n] = (_Float16)v;
        else       ((float*)outv)[(size_t)m * N + n] = v;
      }
    }
}

// ---------------------------------------------------------------------------
// Flash attention: one wave per (b, h, 16-query block). qkv f16 [rows, 3C],
// layout per row: [q(512) | k(512) | v(512)], head slice h*64 (=128B rows).
// ---------------------------------------------------------------------------
__global__ __launch_bounds__(32) void attn_kernel(
    const _Float16* __restrict__ qkv, _Float16* __restrict__ out)
{
  __shared__ __align__(16) _Float16 ldsP[16 * 32];
  __shared__ __align__(16) _Float16 ldsVT[64 * 32];
  const int lane = threadIdx.x & 31;
  const int hi = lane >> 4, l15 = lane & 15;
  const int qb = blockIdx.x * 16;
  const int bh = blockIdx.y;
  const int b = bh >> 3, h = bh & 7;
  const int RS = 3 * CDIM;  // 1536 halves per row
  const _Float16* base = qkv + (size_t)b * TSEQ * RS;
  const int qcol = h * HD, kcol = CDIM + h * HD, vcol = 2 * CDIM + h * HD;

  // Q fragments (A-layout, D split 0..31 / 32..63), direct b128 global loads
  v16h qf[2];
  {
    const _Float16* qp = base + (size_t)(qb + l15) * RS + qcol;
#pragma unroll
    for (int d = 0; d < 2; ++d) {
      HF16 f; const int bk = (hi ? 8 : 0) + d * 32;
      f.q[0] = *(const uint4*)(qp + bk);
      f.q[1] = *(const uint4*)(qp + bk + 16);
      qf[d] = f.v;
    }
  }
  v8f o[4];
#pragma unroll
  for (int j = 0; j < 4; ++j) o[j] = v8f_zero();
  float mo[8], ls[8];
#pragma unroll
  for (int r = 0; r < 8; ++r) { mo[r] = -1.0e30f; ls[r] = 0.0f; }

  const int qmax = qb + 15;
  for (int jb = 0; jb <= qmax; jb += 32) {
    if (jb + 32 <= qmax) {  // warm caches for next key/value block (128B rows)
      __builtin_prefetch(base + (size_t)(jb + 32 + l15) * RS + kcol, 0, 1);
      __builtin_prefetch(base + (size_t)(jb + 48 + l15) * RS + kcol, 0, 1);
      __builtin_prefetch(base + (size_t)(jb + 32 + lane) * RS + vcol, 0, 1);
    }
    // S = Q @ K^T : two 16x16 key tiles, each two k=32 WMMA steps over D
    v8f s[2]; s[0] = v8f_zero(); s[1] = v8f_zero();
#pragma unroll
    for (int t = 0; t < 2; ++t) {
      const _Float16* kp = base + (size_t)(jb + t * 16 + l15) * RS + kcol;
#pragma unroll
      for (int d = 0; d < 2; ++d) {
        HF16 f; const int bk = (hi ? 8 : 0) + d * 32;   // B-layout over D
        f.q[0] = *(const uint4*)(kp + bk);
        f.q[1] = *(const uint4*)(kp + bk + 16);
        s[t] = __builtin_amdgcn_wmma_f32_16x16x32_f16(
            false, qf[d], false, f.v, (short)0, s[t], false, false);
      }
    }
    // online softmax: row r of this wave-half lives across its 16 lanes
#pragma unroll
    for (int r = 0; r < 8; ++r) {
      const int m = qb + r + hi * 8;
      const int key0 = jb + l15, key1 = jb + 16 + l15;
      const float v0 = (key0 <= m) ? s[0][r] * 0.125f : -1.0e30f;
      const float v1 = (key1 <= m) ? s[1][r] * 0.125f : -1.0e30f;
      float rm = fmaxf(v0, v1);
      rm = fmaxf(rm, __shfl_xor(rm, 1));
      rm = fmaxf(rm, __shfl_xor(rm, 2));
      rm = fmaxf(rm, __shfl_xor(rm, 4));
      rm = fmaxf(rm, __shfl_xor(rm, 8));
      const float mn = fmaxf(mo[r], rm);
      const float a  = __expf(mo[r] - mn);
      const float e0 = __expf(v0 - mn);
      const float e1 = __expf(v1 - mn);
      float rs = e0 + e1;
      rs += __shfl_xor(rs, 1);
      rs += __shfl_xor(rs, 2);
      rs += __shfl_xor(rs, 4);
      rs += __shfl_xor(rs, 8);
      ls[r] = ls[r] * a + rs;
      mo[r] = mn;
#pragma unroll
      for (int j = 0; j < 4; ++j) o[j][r] *= a;
      ldsP[(r + hi * 8) * 32 + l15]      = (_Float16)e0;  // C-layout -> LDS
      ldsP[(r + hi * 8) * 32 + 16 + l15] = (_Float16)e1;
    }
    // V block (32 keys x 64 dims) -> LDS transposed [d][key]
    {
      const _Float16* vp = base + (size_t)(jb + lane) * RS + vcol;
#pragma unroll
      for (int dq = 0; dq < 8; ++dq) {
        uint4 u = *(const uint4*)(vp + dq * 8);
        const _Float16* hs = (const _Float16*)&u;
#pragma unroll
        for (int e = 0; e < 8; ++e) ldsVT[(dq * 8 + e) * 32 + lane] = hs[e];
      }
    }
    __syncthreads();
    // P A-fragment from LDS, then O += P @ V (4 d-tiles)
    v16h pf;
    {
      const _Float16* pp = &ldsP[l15 * 32];
      HF16 f; const int bk = hi ? 8 : 0;
      f.q[0] = *(const uint4*)(pp + bk);
      f.q[1] = *(const uint4*)(pp + bk + 16);
      pf = f.v;
    }
#pragma unroll
    for (int j = 0; j < 4; ++j) {
      const _Float16* vtp = &ldsVT[(j * 16 + l15) * 32];
      HF16 f; const int bk = hi ? 8 : 0;
      f.q[0] = *(const uint4*)(vtp + bk);
      f.q[1] = *(const uint4*)(vtp + bk + 16);
      o[j] = __builtin_amdgcn_wmma_f32_16x16x32_f16(
          false, pf, false, f.v, (short)0, o[j], false, false);
    }
    __syncthreads();
  }
  // normalize and store f16 [row, h*64 + d]
#pragma unroll
  for (int r = 0; r < 8; ++r) {
    const float inv = 1.0f / ls[r];
    const size_t orow = (size_t)(b * TSEQ + qb + r + hi * 8) * CDIM + h * HD;
#pragma unroll
    for (int j = 0; j < 4; ++j)
      out[orow + j * 16 + l15] = (_Float16)(o[j][r] * inv);
  }
}

// ---------------------------------------------------------------------------
// Per-row log-softmax NLL over V=32000, then deterministic mean reduction
// ---------------------------------------------------------------------------
__global__ __launch_bounds__(256) void lsm_kernel(
    const float* __restrict__ logits, const int* __restrict__ tgt,
    float* __restrict__ nll)
{
  __shared__ float red[256];
  const int row = blockIdx.x, tid = threadIdx.x;
  const float* lr = logits + (size_t)row * VOC;
  float mx = -3.0e38f;
  for (int i = tid; i < VOC; i += 256) mx = fmaxf(mx, lr[i]);
  red[tid] = mx; __syncthreads();
  for (int s = 128; s > 0; s >>= 1) { if (tid < s) red[tid] = fmaxf(red[tid], red[tid + s]); __syncthreads(); }
  mx = red[0]; __syncthreads();
  float sum = 0.0f;
  for (int i = tid; i < VOC; i += 256) sum += __expf(lr[i] - mx);
  red[tid] = sum; __syncthreads();
  for (int s = 128; s > 0; s >>= 1) { if (tid < s) red[tid] += red[tid + s]; __syncthreads(); }
  if (tid == 0) nll[row] = logf(red[0]) + mx - lr[tgt[row]];
}

__global__ __launch_bounds__(256) void loss_reduce_kernel(
    const float* __restrict__ nll, float* __restrict__ loss)
{
  __shared__ float red[256];
  const int tid = threadIdx.x;
  float s = 0.0f;
  for (int i = tid; i < MROWS; i += 256) s += nll[i];
  red[tid] = s; __syncthreads();
  for (int k = 128; k > 0; k >>= 1) { if (tid < k) red[tid] += red[tid + k]; __syncthreads(); }
  if (tid == 0) loss[0] = red[0] * (1.0f / MROWS);
}

// ---------------------------------------------------------------------------
extern "C" void kernel_launch(void* const* d_in, const int* in_sizes, int n_in,
                              void* d_out, int out_size, void* d_ws, size_t ws_size,
                              hipStream_t stream)
{
  (void)in_sizes; (void)n_in; (void)out_size; (void)ws_size;
  const int*   idx     = (const int*)  d_in[0];
  const int*   targets = (const int*)  d_in[1];
  const float* tok     = (const float*)d_in[2];
  const float* pos     = (const float*)d_in[3];
  const float* qkv_w   = (const float*)d_in[4];
  const float* qkv_b   = (const float*)d_in[5];
  const float* out_w   = (const float*)d_in[6];
  const float* out_b   = (const float*)d_in[7];
  const float* ln1_w   = (const float*)d_in[8];
  const float* ln1_b   = (const float*)d_in[9];
  const float* ff1_w   = (const float*)d_in[10];
  const float* ff1_b   = (const float*)d_in[11];
  const float* ff2_w   = (const float*)d_in[12];
  const float* ff2_b   = (const float*)d_in[13];
  const float* ln2_w   = (const float*)d_in[14];
  const float* ln2_b   = (const float*)d_in[15];
  const float* lnf_w   = (const float*)d_in[16];
  const float* lnf_b   = (const float*)d_in[17];
  const float* head_w  = (const float*)d_in[18];
  const float* head_b  = (const float*)d_in[19];

  // workspace layout (bytes)
  char* ws = (char*)d_ws;
  float*    x   = (float*)   (ws + 0);          // 4096*512  f32  (8 MB)
  _Float16* xn  = (_Float16*)(ws + 8388608);    // 4096*512  f16  (4 MB)
  _Float16* qkv = (_Float16*)(ws + 12582912);   // 4096*1536 f16  (12 MB)
  _Float16* ao  = (_Float16*)(ws + 25165824);   // 4096*512  f16  (4 MB)
  _Float16* h1  = (_Float16*)(ws + 29360128);   // 4096*2048 f16  (16 MB)
  float*    nll = (float*)   (ws + 46137344);   // 4096 f32

  float* logits = (float*)d_out;
  float* loss   = logits + (size_t)MROWS * VOC;

  embed_kernel<<<dim3(MROWS), dim3(256), 0, stream>>>(idx, tok, pos, x);

  for (int l = 0; l < NL; ++l) {
    ln_kernel<<<dim3(MROWS), dim3(256), 0, stream>>>(
        x, ln1_w + (size_t)l * CDIM, ln1_b + (size_t)l * CDIM, xn);
    gemm_kernel<false, false, true><<<dim3((3 * CDIM) / 128, MROWS / 128), dim3(256), 0, stream>>>(
        xn, qkv_w + (size_t)l * 3 * CDIM * CDIM, qkv_b + (size_t)l * 3 * CDIM,
        qkv, nullptr, MROWS, 3 * CDIM, CDIM);
    attn_kernel<<<dim3(TSEQ / 16, NBATCH * NH), dim3(32), 0, stream>>>(qkv, ao);
    gemm_kernel<false, true, false><<<dim3(CDIM / 128, MROWS / 128), dim3(256), 0, stream>>>(
        ao, out_w + (size_t)l * CDIM * CDIM, out_b + (size_t)l * CDIM,
        x, x, MROWS, CDIM, CDIM);
    ln_kernel<<<dim3(MROWS), dim3(256), 0, stream>>>(
        x, ln2_w + (size_t)l * CDIM, ln2_b + (size_t)l * CDIM, xn);
    gemm_kernel<true, false, true><<<dim3((4 * CDIM) / 128, MROWS / 128), dim3(256), 0, stream>>>(
        xn, ff1_w + (size_t)l * 4 * CDIM * CDIM, ff1_b + (size_t)l * 4 * CDIM,
        h1, nullptr, MROWS, 4 * CDIM, CDIM);
    gemm_kernel<false, true, false><<<dim3(CDIM / 128, MROWS / 128), dim3(256), 0, stream>>>(
        h1, ff2_w + (size_t)l * CDIM * 4 * CDIM, ff2_b + (size_t)l * CDIM,
        x, x, MROWS, CDIM, 4 * CDIM);
  }

  ln_kernel<<<dim3(MROWS), dim3(256), 0, stream>>>(x, lnf_w, lnf_b, xn);
  gemm_kernel<false, false, false><<<dim3(VOC / 128, MROWS / 128), dim3(256), 0, stream>>>(
      xn, head_w, head_b, logits, nullptr, MROWS, VOC, CDIM);

  lsm_kernel<<<dim3(MROWS), dim3(256), 0, stream>>>(logits, targets, nll);
  loss_reduce_kernel<<<dim3(1), dim3(256), 0, stream>>>(nll, loss);
}
